// Policy_9543417331947
// MI455X (gfx1250) — matrix operand order repaired and checked by reading, not compiled, not run on hardware
//
#include <hip/hip_runtime.h>
#include <hip/hip_bf16.h>

// Problem dims (match reference config)
#define LNUM  4
#define GNUM  8
#define HDIM  2048
#define RDIM  128
#define SDIM  512
#define ADIM  4096
#define BATCH 4096

typedef __bf16 v16bf __attribute__((ext_vector_type(16)));
typedef __bf16 v8bf  __attribute__((ext_vector_type(8)));
typedef __bf16 v4bf  __attribute__((ext_vector_type(4)));
typedef float  v8f   __attribute__((ext_vector_type(8)));

constexpr int BM = 128, BN = 128, BK = 64;   // 2 WMMA k-steps per LDS stage
constexpr int BKP = BK + 8;                  // pad rows to 144B (16B-aligned), dodge bank conflicts
constexpr int MAXMT = BATCH / BM;            // 32 m-tiles worst case per expert

// ---------------------------------------------------------------------------
// Routing: build per-expert compacted row index lists
// ---------------------------------------------------------------------------
__global__ void k_route_init(int* counts) {
    if (threadIdx.x < GNUM) counts[threadIdx.x] = 0;
}
__global__ void k_route_count(const int* __restrict__ o, int* __restrict__ counts) {
    int b = blockIdx.x * 256 + threadIdx.x;
    if (b < BATCH) atomicAdd(&counts[o[b]], 1);
}
__global__ void k_route_scan(const int* __restrict__ counts, int* __restrict__ offs,
                             int* __restrict__ cursor) {
    if (threadIdx.x == 0) {
        int a = 0;
        for (int g = 0; g < GNUM; ++g) { offs[g] = a; cursor[g] = a; a += counts[g]; }
        offs[GNUM] = a;
    }
}
__global__ void k_route_scatter(const int* __restrict__ o, int* __restrict__ cursor,
                                int* __restrict__ idx) {
    int b = blockIdx.x * 256 + threadIdx.x;
    if (b < BATCH) { int p = atomicAdd(&cursor[o[b]], 1); idx[p] = b; }
}

// ---------------------------------------------------------------------------
// Tiled bf16-WMMA GEMM, fp32 in/out, on-the-fly fp32->bf16 conversion,
// software-pipelined: next tile is fetched into registers while WMMAs run
// on the current LDS stage (hides HBM latency; workload is BW-bound).
// MODE 0: dense          C[i]      = A[i]      @ B + bias
// MODE 1: gather-A       Ccomp[i]  = A[idx[i]] @ B_g + bias_g   (expert down-proj)
// MODE 2: scatter-C+ELU  C[idx[i]] = elu(add[idx[i]] + Acomp[i] @ B_g + bias_g)
// Block tile 128x128x64; 8 waves; wave tile 32x64 (2x4 WMMA 16x16x32 bf16, 2 k-steps).
// ---------------------------------------------------------------------------
template <int MODE>
__global__ __launch_bounds__(256)
void gemm_wmma(const float* __restrict__ A, int lda,
               const float* __restrict__ Bm, int ldb,
               const float* __restrict__ bias,
               float* __restrict__ C, int ldc,
               int N, int K,
               const int* __restrict__ idx,
               const int* __restrict__ offs,
               const int* __restrict__ counts,
               long strideBg,
               const float* __restrict__ addsrc, int ldadd)
{
    __shared__ __bf16 As[BM][BKP];
    __shared__ __bf16 Bs[BN][BKP];

    const int lane = threadIdx.x & 31;
    const int wave = threadIdx.x >> 5;
    const int wm = wave & 3;    // 0..3 : 32-row strip
    const int wn = wave >> 2;   // 0..1 : 64-col strip

    int rowBeg, rowEnd;
    if constexpr (MODE == 0) {
        rowBeg = blockIdx.x * BM;
        rowEnd = rowBeg + BM;
    } else {
        int g  = blockIdx.x / MAXMT;
        int mt = blockIdx.x % MAXMT;
        int base = offs[g], cnt = counts[g];
        rowBeg = base + mt * BM;
        rowEnd = base + cnt;
        if (rowBeg >= rowEnd) return;
        Bm   += (size_t)g * strideBg;   // per-expert weight slab
        bias += (size_t)g * N;          // per-expert bias row
    }
    const int n0 = blockIdx.y * BN;

    // ---- global->LDS work assignment (256 threads) ----
    // A: thread covers 1 row x 32 cols (8 float4)
    const int arl = threadIdx.x >> 1;          // A row 0..127
    const int acb = (threadIdx.x & 1) << 5;    // A col base 0/32
    int ai = rowBeg + arl;
    if constexpr (MODE != 0) { if (ai > rowEnd - 1) ai = rowEnd - 1; }  // clamp padded rows
    int arow;
    if constexpr (MODE == 1) arow = idx[ai];
    else                     arow = ai;
    const float* aptr = A + (size_t)arow * lda + acb;

    // B: thread covers 4 consecutive k-rows x 8 cols (8 float4) -> packs 4 bf16
    // along K per column so the LDS transpose uses ds_store_b64 instead of b16.
    const int bk4  = (threadIdx.x >> 4) << 2;  // k base 0,4,...,60
    const int bcb8 = (threadIdx.x & 15) << 3;  // col base 0..120 step 8
    const float* bbase = Bm + (size_t)bk4 * ldb + n0 + bcb8;

    const int mr   = lane & 15;
    const int half = lane >> 4;

    v8f acc[2][4] = {};

    const int KT = K / BK;

    // ---- prologue: fetch tile 0 into registers ----
    float4 aReg[8], bReg[8];
    #pragma unroll
    for (int j = 0; j < 8; ++j) aReg[j] = ((const float4*)aptr)[j];
    #pragma unroll
    for (int j = 0; j < 8; ++j)
        bReg[j] = ((const float4*)(bbase + (size_t)(j >> 1) * ldb))[j & 1];

    for (int kt = 0; kt < KT; ++kt) {
        // ---- convert current registers -> LDS stage ----
        {
            alignas(16) __bf16 ta[32];
            #pragma unroll
            for (int j = 0; j < 8; ++j) {
                float4 f = aReg[j];
                ta[4*j+0] = (__bf16)f.x; ta[4*j+1] = (__bf16)f.y;
                ta[4*j+2] = (__bf16)f.z; ta[4*j+3] = (__bf16)f.w;
            }
            #pragma unroll
            for (int c = 0; c < 4; ++c)
                *(v8bf*)&As[arl][acb + 8*c] = *(const v8bf*)&ta[8*c];
        }
        #pragma unroll
        for (int e = 0; e < 8; ++e) {
            v4bf q;
            #pragma unroll
            for (int r = 0; r < 4; ++r) {
                float4 f = bReg[r*2 + (e >> 2)];
                float fv = ((e & 3) == 0) ? f.x : ((e & 3) == 1) ? f.y
                         : ((e & 3) == 2) ? f.z : f.w;
                q[r] = (__bf16)fv;
            }
            *(v4bf*)&Bs[bcb8 + e][bk4] = q;
        }
        __syncthreads();

        // ---- issue next tile's global loads (overlap with WMMA below) ----
        if (kt + 1 < KT) {
            const float* an = aptr + (kt + 1) * BK;
            const float* bn = bbase + (size_t)(kt + 1) * BK * ldb;
            #pragma unroll
            for (int j = 0; j < 8; ++j) aReg[j] = ((const float4*)an)[j];
            #pragma unroll
            for (int j = 0; j < 8; ++j)
                bReg[j] = ((const float4*)(bn + (size_t)(j >> 1) * ldb))[j & 1];
            if (kt + 2 < KT) {   // pull next-next weight/act lines toward GL2
                __builtin_prefetch(aptr + (kt + 2) * BK, 0, 0);
                __builtin_prefetch(bbase + (size_t)(kt + 2) * BK * ldb, 0, 0);
            }
        }

        // ---- 2 WMMA k-steps from LDS ----
        #pragma unroll
        for (int ks = 0; ks < 2; ++ks) {
            const int kb = ks * 32;
            // B fragments: lane = column N=lane%16, K-half by lane/16 (K packed 2/VGPR)
            v16bf bfrag[4];
            #pragma unroll
            for (int tn = 0; tn < 4; ++tn) {
                const __bf16* p = &Bs[wn * 64 + tn * 16 + mr][kb + half * 16];
                v8bf lo = *(const v8bf*)p;
                v8bf hi = *(const v8bf*)(p + 8);
                bfrag[tn] = __builtin_shufflevector(lo, hi, 0,1,2,3,4,5,6,7,8,9,10,11,12,13,14,15);
            }
            // A fragments: lane = row M=lane%16; K = 16*half..+15
            #pragma unroll
            for (int tm = 0; tm < 2; ++tm) {
                const __bf16* pa = &As[wm * 32 + tm * 16 + mr][kb];
                v8bf lo = *(const v8bf*)(pa + 16 * half);
                v8bf hi = *(const v8bf*)(pa + 16 * half + 8);
                v16bf afrag = __builtin_shufflevector(lo, hi, 0,1,2,3,4,5,6,7,8,9,10,11,12,13,14,15);
                #pragma unroll
                for (int tn = 0; tn < 4; ++tn)
                    acc[tm][tn] = __builtin_amdgcn_wmma_f32_16x16x32_bf16(
                        false, afrag, false, bfrag[tn], (short)0, acc[tm][tn], false, false);
            }
        }
        __syncthreads();
    }

    // ---- epilogue: C/D layout -> element (vgpr v, lane L): M = v + 8*(L/16), N = L%16
    #pragma unroll
    for (int tm = 0; tm < 2; ++tm) {
        #pragma unroll
        for (int tn = 0; tn < 4; ++tn) {
            int col = n0 + wn * 64 + tn * 16 + mr;
            float bv = bias[col];
            #pragma unroll
            for (int v = 0; v < 8; ++v) {
                int rloc = wm * 32 + tm * 16 + half * 8 + v;
                int i = rowBeg + rloc;
                if constexpr (MODE != 0) { if (i >= rowEnd) continue; }
                float val = acc[tm][tn][v] + bv;
                if constexpr (MODE == 2) {
                    int crow = idx[i];
                    float sum = addsrc[(size_t)crow * ldadd + col] + val;
                    C[(size_t)crow * ldc + col] = (sum > 0.f) ? sum : (__expf(sum) - 1.f);
                } else {
                    C[(size_t)i * ldc + col] = val;   // MODE 0: dense row; MODE 1: compact row
                }
            }
        }
    }
}

// ---------------------------------------------------------------------------
extern "C" void kernel_launch(void* const* d_in, const int* in_sizes, int n_in,
                              void* d_out, int out_size, void* d_ws, size_t ws_size,
                              hipStream_t stream)
{
    const float* s      = (const float*)d_in[0];
    const int*   o      = (const int*)  d_in[1];
    const float* W_in   = (const float*)d_in[2];
    const float* b_in   = (const float*)d_in[3];
    const float* W_main = (const float*)d_in[4];
    const float* b_main = (const float*)d_in[5];
    const float* W_g1   = (const float*)d_in[6];
    const float* b_g1   = (const float*)d_in[7];
    const float* W_g2   = (const float*)d_in[8];
    const float* b_g2   = (const float*)d_in[9];
    const float* W_out  = (const float*)d_in[10];
    const float* b_out  = (const float*)d_in[11];
    float* out = (float*)d_out;

    char* p = (char*)d_ws;
    float* x    = (float*)p;  p += (size_t)BATCH * HDIM * sizeof(float);
    float* acc  = (float*)p;  p += (size_t)BATCH * HDIM * sizeof(float);
    float* h1c  = (float*)p;  p += (size_t)BATCH * RDIM * sizeof(float);
    int* idx    = (int*)p;    p += BATCH * sizeof(int);
    int* counts = (int*)p;    p += 64 * sizeof(int);
    int* offs   = (int*)p;    p += 64 * sizeof(int);
    int* cursor = (int*)p;    p += 64 * sizeof(int);
    (void)ws_size; (void)in_sizes; (void)n_in; (void)out_size;

    // Routing compaction
    k_route_init   <<<1, 32, 0, stream>>>(counts);
    k_route_count  <<<BATCH / 256, 256, 0, stream>>>(o, counts);
    k_route_scan   <<<1, 1, 0, stream>>>(counts, offs, cursor);
    k_route_scatter<<<BATCH / 256, 256, 0, stream>>>(o, cursor, idx);

    dim3 blk(256);
    // x = s @ W_in + b_in   (M=B, N=H, K=S)
    gemm_wmma<0><<<dim3(BATCH / BM, HDIM / BN), blk, 0, stream>>>(
        s, SDIM, W_in, HDIM, b_in, x, HDIM, HDIM, SDIM,
        nullptr, nullptr, nullptr, 0, nullptr, 0);

    for (int l = 0; l < LNUM; ++l) {
        // acc = x @ W_main[l] + b_main[l]
        gemm_wmma<0><<<dim3(BATCH / BM, HDIM / BN), blk, 0, stream>>>(
            x, HDIM, W_main + (size_t)l * HDIM * HDIM, HDIM,
            b_main + (size_t)l * HDIM, acc, HDIM, HDIM, HDIM,
            nullptr, nullptr, nullptr, 0, nullptr, 0);
        // h1c[comp] = x[idx] @ W_g1[l,g] + b_g1[l,g]
        gemm_wmma<1><<<dim3(GNUM * MAXMT, RDIM / BN), blk, 0, stream>>>(
            x, HDIM, W_g1 + (size_t)l * GNUM * HDIM * RDIM, RDIM,
            b_g1 + (size_t)l * GNUM * RDIM, h1c, RDIM, RDIM, HDIM,
            idx, offs, counts, (long)HDIM * RDIM, nullptr, 0);
        // x[idx] = elu(acc[idx] + h1c[comp] @ W_g2[l,g] + b_g2[l,g])
        gemm_wmma<2><<<dim3(GNUM * MAXMT, HDIM / BN), blk, 0, stream>>>(
            h1c, RDIM, W_g2 + (size_t)l * GNUM * RDIM * HDIM, HDIM,
            b_g2 + (size_t)l * GNUM * HDIM, x, HDIM, HDIM, RDIM,
            idx, offs, counts, (long)RDIM * HDIM, acc, HDIM);
    }

    // out = x @ W_out + b_out   (M=B, N=A, K=H)
    gemm_wmma<0><<<dim3(BATCH / BM, ADIM / BN), blk, 0, stream>>>(
        x, HDIM, W_out, ADIM, b_out, out, ADIM, ADIM, HDIM,
        nullptr, nullptr, nullptr, 0, nullptr, 0);
}
